// Model_87351044866495
// MI455X (gfx1250) — matrix operand order repaired and checked by reading, not compiled, not run on hardware
//
#include <hip/hip_runtime.h>
#include <hip/hip_bf16.h>

typedef __attribute__((ext_vector_type(16))) _Float16 v16h;
typedef __attribute__((ext_vector_type(8)))  _Float16 v8h;
typedef __attribute__((ext_vector_type(4)))  float    v4f;
typedef __attribute__((ext_vector_type(8)))  float    v8f;

#define INV_SQRT_1P_EPS 0.99999500003749981f   // 1/sqrt(1+1e-5)

static inline int PPAD(int p) { return (p + 15) & ~15; }   // 64B-aligned pitch

// ---------------------------------------------------------------- prep kernels
__global__ void cvt_pad_f16_kernel(const float* __restrict__ s, _Float16* __restrict__ d,
                                   int O, int C, int Cpad) {
    int i = blockIdx.x * blockDim.x + threadIdx.x;
    if (i >= O * Cpad) return;
    int o = i / Cpad, k = i - o * Cpad;
    d[i] = (k < C) ? (_Float16)s[o * C + k] : (_Float16)0.0f;
}

__global__ void make_ai_kernel(const float* __restrict__ A, const float* __restrict__ ei,
                               float* __restrict__ Ai, int n) {
    int i = blockIdx.x * blockDim.x + threadIdx.x;
    if (i < n) Ai[i] = A[i % 1875] * ei[i];
}

// x (N,3,T,25,1) flat -> pitched (N,3,[PP]) with data-BN applied
__global__ void data_bn_kernel(const float* __restrict__ x, const float* __restrict__ g,
                               const float* __restrict__ b, float* __restrict__ out,
                               int total, int T, int PPx) {
    int i = blockIdx.x * blockDim.x + threadIdx.x;
    if (i >= total) return;
    int col  = i % (25 * T);
    int v    = i % 25;
    int nc   = i / (25 * T);
    int c    = nc % 3;
    int ch   = v * 3 + c;
    out[(long long)nc * PPx + col] = x[i] * (g[ch] * INV_SQRT_1P_EPS) + b[ch];
}

// zero the leading 128 floats and trailing [128+P, ZPP) of each Z plane
__global__ void zero_halo_kernel(float* __restrict__ Z, int planes, int ZPP, int P) {
    int per = ZPP - P;
    int i = blockIdx.x * blockDim.x + threadIdx.x;
    if (i >= planes * per) return;
    int p = i / per, r = i - p * per;
    int off = (r < 128) ? r : (P + r);
    Z[(long long)p * ZPP + off] = 0.f;
}

// --------------------------------------------------------- WMMA fragment glue
// ISA 7.12.2: A(16x32) lane: M=lane&15, half j -> K=16*(j/8)+8*(lane/16)+(j%8)
// B(32x16): N=lane&15, K=16*(j/8)+8*(lane/16)+(j%8).  D: M=8*(lane/16)+r, N=lane&15.
__device__ __forceinline__ v16h frag_cat(v8h lo, v8h hi) {
    return __builtin_shufflevector(lo, hi, 0,1,2,3,4,5,6,7,8,9,10,11,12,13,14,15);
}

// ------------------------------------------------ GCN pointwise GEMM via LDS
// Block 256 = 8 waves (2x4), tile 64(O) x 64(cols), 2 WMMA/wave per K-step.
// Output Y stored in f16 (consumed only by the WMMA aggregation A-operand).
__global__ __launch_bounds__(256) void gcn_gemm_lds_kernel(
    const _Float16* __restrict__ W,   // [O][Cpad] f16, zero padded
    const float*   __restrict__ X,    // chunk base, [nb][C][PPin]
    _Float16*      __restrict__ Out,  // [nb][O][PPin] f16
    const float*   __restrict__ bias,
    int O, int C, int Cpad, int Pout, int PPin,
    long long x_nstride, long long out_nstride)
{
    __shared__ _Float16 As[64 * 40];
    __shared__ _Float16 Bs[64 * 40];
    const int tid    = threadIdx.x;
    const int lane   = tid & 31, wid = tid >> 5;
    const int waveO  = wid >> 2, waveC = wid & 3;
    const int laneHi = lane >> 4, laneLo = lane & 15;
    const int nb     = blockIdx.z;
    const int oBase  = blockIdx.y * 64;
    const int cBase  = blockIdx.x * 64;
    const float* Xn  = X + (long long)nb * x_nstride;

    const int arow = tid >> 2, aseg = tid & 3;
    const int bk   = tid >> 3;
    const int bc0  = (tid & 7) * 8;

    v8f acc0 = {0.f,0.f,0.f,0.f,0.f,0.f,0.f,0.f};
    v8f acc1 = {0.f,0.f,0.f,0.f,0.f,0.f,0.f,0.f};
    for (int kk = 0; kk < Cpad; kk += 32) {
        *(v8h*)(As + arow * 40 + aseg * 8) =
            *(const v8h*)(W + (long long)(oBase + arow) * Cpad + kk + aseg * 8);
        const int krow = kk + bk;
        const float* src = Xn + (long long)krow * PPin + cBase + bc0;
        v4f x0 = *(const v4f*)(src);
        v4f x1 = *(const v4f*)(src + 4);
#pragma unroll
        for (int i = 0; i < 8; ++i) {
            float xv = (i < 4) ? x0[i & 3] : x1[i & 3];
            xv = (krow < C) ? xv : 0.f;
            Bs[(bc0 + i) * 40 + bk] = (_Float16)xv;
        }
        __syncthreads();
        const _Float16* ap = As + (waveO * 32 + laneLo) * 40 + laneHi * 8;
        const _Float16* bp = Bs + (waveC * 16 + laneLo) * 40 + laneHi * 8;
        v16h af0 = frag_cat(*(const v8h*)ap, *(const v8h*)(ap + 16));
        v16h af1 = frag_cat(*(const v8h*)(ap + 16 * 40), *(const v8h*)(ap + 16 * 40 + 16));
        v16h bf  = frag_cat(*(const v8h*)bp, *(const v8h*)(bp + 16));
        acc0 = __builtin_amdgcn_wmma_f32_16x16x32_f16(false, af0, false, bf, (short)0, acc0, false, false);
        acc1 = __builtin_amdgcn_wmma_f32_16x16x32_f16(false, af1, false, bf, (short)0, acc1, false, false);
        __syncthreads();
    }

    const int colB = cBase + waveC * 16 + laneLo;
    if (colB >= Pout) return;
    _Float16* On = Out + (long long)nb * out_nstride;
#pragma unroll
    for (int r = 0; r < 8; ++r) {
        int o0 = oBase + waveO * 32 + (laneHi << 3) + r;
        On[(long long)o0 * PPin + colB] = (_Float16)(acc0[r] + bias[o0]);
        int o1 = o0 + 16;
        On[(long long)o1 * PPin + colB] = (_Float16)(acc1[r] + bias[o1]);
    }
}

// ----------------------------- graph aggregation on WMMA + BN1 + ReLU
// z[o, t, w] = sum_k sum_v Y[(k,o),(t,v)] * Ai[k,v,w]
// Per wave: one (o-tile, t-group of 5). A = 16(o) x 32(v pad), B = Ai_k (32x32 pad, 2 N-tiles),
// accumulate 3 k's into the same acc -> 6 WMMAs per wave between epilogues (30 per wave in all).
__global__ __launch_bounds__(256) void agg_wmma_kernel(
    const _Float16* __restrict__ Y,   // [nb][3*cout][PPy] f16
    const float*   __restrict__ Ai,   // [3][25][25] (layer slice)
    const float*   __restrict__ g1,
    const float*   __restrict__ be1,
    float*         __restrict__ Z,    // [nb][cout][ZPP], interior at +128
    int cout, int PPy, int ZPP, int nTiles)
{
    const int lane   = threadIdx.x & 31, wid = threadIdx.x >> 5;
    const int laneHi = lane >> 4, laneLo = lane & 15;
    const int nb     = blockIdx.y;
    const int unit   = blockIdx.x * 8 + wid;      // exact: (T/5)*nTiles % 8 == 0
    const int tg     = unit / nTiles;
    const int otile  = unit - tg * nTiles;
    const int t0     = tg * 5;
    const int oBase  = otile * 16;

    // B fragments: bf[k][tile], K=v=16*(j/8)+8*laneHi+(j%8), N=w=laneLo+16*tile
    v16h bf[3][2];
#pragma unroll
    for (int k = 0; k < 3; ++k)
#pragma unroll
        for (int tile = 0; tile < 2; ++tile) {
            int w = laneLo + tile * 16;
#pragma unroll
            for (int j = 0; j < 16; ++j) {
                int v = ((j >> 3) << 4) + laneHi * 8 + (j & 7);
                float av = Ai[k * 625 + v * 25 + w];     // slack-padded buffer
                bf[k][tile][j] = (_Float16)((v < 25 && w < 25) ? av : 0.f);
            }
        }
    // epilogue constants for the 8 owned rows
    float sg[8], sb[8];
#pragma unroll
    for (int r = 0; r < 8; ++r) {
        int o = oBase + (laneHi << 3) + r;
        sg[r] = g1[o] * INV_SQRT_1P_EPS;
        sb[r] = be1[o];
    }

    const long long yrow = ((long long)nb * 3 * cout + oBase + laneLo) * PPy;
    for (int tt = 0; tt < 5; ++tt) {
        const int t = t0 + tt;
        v8f a0 = {0.f,0.f,0.f,0.f,0.f,0.f,0.f,0.f};
        v8f a1 = {0.f,0.f,0.f,0.f,0.f,0.f,0.f,0.f};
#pragma unroll
        for (int k = 0; k < 3; ++k) {
            const _Float16* Yrow = Y + yrow + (long long)k * cout * PPy + t * 25;
            v16h af;
#pragma unroll
            for (int j = 0; j < 8; ++j) {
                af[j] = Yrow[8 * laneHi + j];            // v<=15, always valid
                int v1 = 16 + 8 * laneHi + j;
                _Float16 hv = Yrow[v1];                  // in-slack over-read
                af[j + 8] = (v1 < 25) ? hv : (_Float16)0.0f;
            }
            a0 = __builtin_amdgcn_wmma_f32_16x16x32_f16(false, af, false, bf[k][0], (short)0, a0, false, false);
            a1 = __builtin_amdgcn_wmma_f32_16x16x32_f16(false, af, false, bf[k][1], (short)0, a1, false, false);
        }
        float* Zp = Z + ((long long)nb * cout + oBase + (laneHi << 3)) * ZPP + 128 + t * 25;
#pragma unroll
        for (int r = 0; r < 8; ++r) {
            float* Zr = Zp + (long long)r * ZPP;
            Zr[laneLo] = fmaxf(a0[r] * sg[r] + sb[r], 0.f);
            int w1 = 16 + laneLo;
            if (w1 < 25) Zr[w1] = fmaxf(a1[r] * sg[r] + sb[r], 0.f);
        }
    }
}

// ---------------------- WMMA implicit-GEMM temporal conv (9x1) via LDS + epi
__global__ __launch_bounds__(256) void tcn_gemm_lds_kernel(
    const _Float16* __restrict__ W,    // [O][R=C*9] f16
    const float*   __restrict__ Z,     // [nb][C][ZPP], interior at +128
    float*         __restrict__ Out,   // chunk base, [nb][O][PPout]
    const float*   __restrict__ resid, // chunk base or null
    const float*   __restrict__ bias,
    const float*   __restrict__ bng,
    const float*   __restrict__ bnb,
    int O, int C, int Pout, int PPout, int ZPP, int V, int tstride,
    long long out_nstride, long long res_nstride)
{
    __shared__ _Float16 As[64 * 40];
    __shared__ _Float16 Bs[64 * 40];
    const int tid    = threadIdx.x;
    const int lane   = tid & 31, wid = tid >> 5;
    const int waveO  = wid >> 2, waveC = wid & 3;
    const int laneHi = lane >> 4, laneLo = lane & 15;
    const int nb     = blockIdx.z;
    const int oBase  = blockIdx.y * 64;
    const int cBase  = blockIdx.x * 64;
    const int R      = C * 9;

    const int arow = tid >> 2, aseg = tid & 3;
    const int bk   = tid >> 3;
    const int bc0  = (tid & 7) * 8;

    v8f acc0 = {0.f,0.f,0.f,0.f,0.f,0.f,0.f,0.f};
    v8f acc1 = {0.f,0.f,0.f,0.f,0.f,0.f,0.f,0.f};
    for (int kk = 0; kk < R; kk += 32) {
        *(v8h*)(As + arow * 40 + aseg * 8) =
            *(const v8h*)(W + (long long)(oBase + arow) * R + kk + aseg * 8);
        const int r  = kk + bk;
        const int cI = r / 9;
        const int dt = r - cI * 9;
        const float* plane = Z + ((long long)nb * C + cI) * ZPP + 128;
        if (tstride == 1) {
            const float* src = plane + cBase + bc0 + (dt - 4) * V;
#pragma unroll
            for (int i = 0; i < 8; ++i)
                Bs[(bc0 + i) * 40 + bk] = (_Float16)src[i];
        } else {
#pragma unroll
            for (int i = 0; i < 8; ++i) {
                int col = cBase + bc0 + i;
                int t = col / V, v = col - t * V;
                Bs[(bc0 + i) * 40 + bk] = (_Float16)plane[(t * tstride + dt - 4) * V + v];
            }
        }
        __syncthreads();
        const _Float16* ap = As + (waveO * 32 + laneLo) * 40 + laneHi * 8;
        const _Float16* bp = Bs + (waveC * 16 + laneLo) * 40 + laneHi * 8;
        v16h af0 = frag_cat(*(const v8h*)ap, *(const v8h*)(ap + 16));
        v16h af1 = frag_cat(*(const v8h*)(ap + 16 * 40), *(const v8h*)(ap + 16 * 40 + 16));
        v16h bf  = frag_cat(*(const v8h*)bp, *(const v8h*)(bp + 16));
        acc0 = __builtin_amdgcn_wmma_f32_16x16x32_f16(false, af0, false, bf, (short)0, acc0, false, false);
        acc1 = __builtin_amdgcn_wmma_f32_16x16x32_f16(false, af1, false, bf, (short)0, acc1, false, false);
        __syncthreads();
    }

    const int colB = cBase + waveC * 16 + laneLo;
    if (colB >= Pout) return;
    float* On = Out + (long long)nb * out_nstride;
#pragma unroll
    for (int r8 = 0; r8 < 8; ++r8) {
        int o0 = oBase + waveO * 32 + (laneHi << 3) + r8;
        int o1 = o0 + 16;
        float v0 = (acc0[r8] + bias[o0]) * (bng[o0] * INV_SQRT_1P_EPS) + bnb[o0];
        float v1 = (acc1[r8] + bias[o1]) * (bng[o1] * INV_SQRT_1P_EPS) + bnb[o1];
        if (resid) {
            const float* Rn = resid + (long long)nb * res_nstride;
            v0 += Rn[(long long)o0 * PPout + colB];
            v1 += Rn[(long long)o1 * PPout + colB];
        }
        On[(long long)o0 * PPout + colB] = fmaxf(v0, 0.f);
        On[(long long)o1 * PPout + colB] = fmaxf(v1, 0.f);
    }
}

// ---------------------- strided residual 1x1 conv (small), direct per-wave WMMA
__global__ void res_gemm_wmma_kernel(
    const _Float16* __restrict__ W,   // [O][C] f16
    const float*   __restrict__ X,    // chunk base, [nb][C][PPin]
    float*         __restrict__ Out,  // [nb][O][PPout]
    const float*   __restrict__ bias,
    const float*   __restrict__ bng,
    const float*   __restrict__ bnb,
    int O, int C, int Pout, int PPout, int PPin, int V, int tstride,
    long long x_nstride, long long out_nstride)
{
    const int lane   = threadIdx.x & 31;
    const int laneHi = lane >> 4;
    const int laneLo = lane & 15;
    const int nb     = blockIdx.z;
    const int otile  = blockIdx.y << 4;
    const int ptile  = blockIdx.x << 4;

    const float* Xn = X + (long long)nb * x_nstride;
    const int  colB = ptile + laneLo;
    const int  colC = (colB < Pout) ? colB : (Pout - 1);
    const int  tB   = colC / V;
    const int  vB   = colC - tB * V;
    const long long boff = (long long)tB * tstride * V + vB;
    const int  aRow = otile + laneLo;

    v8f acc = {0.f,0.f,0.f,0.f,0.f,0.f,0.f,0.f};
    for (int kk = 0; kk < C; kk += 32) {
        v16h af, bf;
#pragma unroll
        for (int j = 0; j < 16; ++j) {
            int ka = kk + ((j >> 3) << 4) + (laneHi << 3) + (j & 7);
            af[j] = W[(long long)aRow * C + ka];
            int kb = kk + ((j >> 3) << 4) + (laneHi << 3) + (j & 7);
            bf[j] = (_Float16)Xn[(long long)kb * PPin + boff];
        }
        acc = __builtin_amdgcn_wmma_f32_16x16x32_f16(false, af, false, bf, (short)0, acc, false, false);
    }
    if (colB >= Pout) return;
    float* On = Out + (long long)nb * out_nstride;
#pragma unroll
    for (int r = 0; r < 8; ++r) {
        int o = otile + (laneHi << 3) + r;
        On[(long long)o * PPout + colB] =
            (acc[r] + bias[o]) * (bng[o] * INV_SQRT_1P_EPS) + bnb[o];
    }
}

// ---------------------------------------------------------------- pool + FC
__global__ void pool_kernel(const float* __restrict__ X, float* __restrict__ P,
                            int Cc, int len, int pitch) {
    int n = blockIdx.x;
    for (int c = threadIdx.x; c < Cc; c += blockDim.x) {
        const float* p = X + ((long long)n * Cc + c) * pitch;
        float s = 0.f;
        for (int i = 0; i < len; ++i) s += p[i];
        P[n * Cc + c] = s / (float)len;
    }
}

__global__ void fc_kernel(const float* __restrict__ P, const float* __restrict__ Wf,
                          const float* __restrict__ bf, float* __restrict__ out,
                          int N, int Cc, int ncls) {
    int i = blockIdx.x * blockDim.x + threadIdx.x;
    if (i >= N * ncls) return;
    int j = i % ncls, n = i / ncls;
    float s = bf[j];
    const float* p = P + (long long)n * Cc;
    const float* w = Wf + (long long)j * Cc;
    for (int c = 0; c < Cc; ++c) s += p[c] * w[c];
    out[i] = s;
}

// ---------------------------------------------------------------- host driver
extern "C" void kernel_launch(void* const* d_in, const int* in_sizes, int n_in,
                              void* d_out, int out_size, void* d_ws, size_t ws_size,
                              hipStream_t stream)
{
    (void)in_sizes; (void)n_in; (void)out_size; (void)ws_size;
    const int N = 128, V = 25, K = 3, T0 = 300, NC = 16;
    struct LC { int cin, cout, stride, resid; };
    static const LC cfg[10] = {{3,64,1,0},{64,64,1,1},{64,64,1,1},{64,64,1,1},
                               {64,128,2,1},{128,128,1,1},{128,128,1,1},
                               {128,256,2,1},{256,256,1,1},{256,256,1,1}};

    const float* x = (const float*)d_in[0];
    const float* A = (const float*)d_in[1];
    int pi = 2;
    const float* dbn_g = (const float*)d_in[pi++];
    const float* dbn_b = (const float*)d_in[pi++];
    const float* ei    = (const float*)d_in[pi++];
    const float *w_gcn[10], *b_gcn[10], *g1[10], *be1[10];
    const float *w_tcn[10], *b_tcn[10], *g2[10], *be2[10];
    const float *w_res[10] = {}, *b_res[10] = {}, *g_res[10] = {}, *be_res[10] = {};
    bool hasres[10];
    for (int i = 0; i < 10; ++i) {
        w_gcn[i] = (const float*)d_in[pi++];
        b_gcn[i] = (const float*)d_in[pi++];
        g1[i]    = (const float*)d_in[pi++];
        be1[i]   = (const float*)d_in[pi++];
        w_tcn[i] = (const float*)d_in[pi++];
        b_tcn[i] = (const float*)d_in[pi++];
        g2[i]    = (const float*)d_in[pi++];
        be2[i]   = (const float*)d_in[pi++];
        hasres[i] = cfg[i].resid && (cfg[i].cin != cfg[i].cout || cfg[i].stride != 1);
        if (hasres[i]) {
            w_res[i]  = (const float*)d_in[pi++];
            b_res[i]  = (const float*)d_in[pi++];
            g_res[i]  = (const float*)d_in[pi++];
            be_res[i] = (const float*)d_in[pi++];
        }
    }
    const float* w_fc = (const float*)d_in[pi++];
    const float* b_fc = (const float*)d_in[pi++];

    char* base = (char*)d_ws; size_t off = 0;
    auto alloc = [&](size_t bytes) -> void* {
        void* p = base + off;
        off = (off + bytes + 8192 + 255) & ~(size_t)255; return p;
    };
    float*    Xa      = (float*)alloc((size_t)62000000 * 4);
    float*    Xb      = (float*)alloc((size_t)62000000 * 4);
    _Float16* Ychunk  = (_Float16*)alloc((size_t)NC * 2900000 * 2);   // f16 now
    float*    Zchunk  = (float*)alloc((size_t)NC * 1100000 * 4);
    float*    Rchunk  = (float*)alloc((size_t)NC * 500000 * 4);
    float*    Aibuf   = (float*)alloc((size_t)10 * 1875 * 4);
    float*    pooled  = (float*)alloc((size_t)N * 256 * 4);
    int cpadG[10];
    _Float16* whg[10]; _Float16* wht[10]; _Float16* whr[10] = {};
    for (int i = 0; i < 10; ++i) {
        cpadG[i] = (cfg[i].cin + 31) & ~31;
        whg[i] = (_Float16*)alloc((size_t)K * cfg[i].cout * cpadG[i] * 2);
        wht[i] = (_Float16*)alloc((size_t)cfg[i].cout * cfg[i].cout * 9 * 2);
        if (hasres[i]) whr[i] = (_Float16*)alloc((size_t)cfg[i].cout * cfg[i].cin * 2);
    }

    auto cvtpad = [&](const float* s, _Float16* d, int O, int C, int Cp) {
        int n = O * Cp;
        cvt_pad_f16_kernel<<<(n + 255) / 256, 256, 0, stream>>>(s, d, O, C, Cp);
    };
    for (int i = 0; i < 10; ++i) {
        cvtpad(w_gcn[i], whg[i], K * cfg[i].cout, cfg[i].cin, cpadG[i]);
        cvtpad(w_tcn[i], wht[i], cfg[i].cout, cfg[i].cout * 9, cfg[i].cout * 9);
        if (hasres[i]) cvtpad(w_res[i], whr[i], cfg[i].cout, cfg[i].cin, cfg[i].cin);
    }
    make_ai_kernel<<<(18750 + 255) / 256, 256, 0, stream>>>(A, ei, Aibuf, 18750);

    const int PP0 = PPAD(T0 * V);
    const int totBN = N * 3 * T0 * V;
    data_bn_kernel<<<(totBN + 255) / 256, 256, 0, stream>>>(x, dbn_g, dbn_b, Xa, totBN, T0, PP0);

    float* Xcur = Xa; float* Xnext = Xb; int Tcur = T0;
    for (int i = 0; i < 10; ++i) {
        const int cin = cfg[i].cin, cout = cfg[i].cout, s = cfg[i].stride;
        const int Tout  = (Tcur - 1) / s + 1;
        const int Pin   = Tcur * V,  PoutT = Tout * V;
        const int PPin  = PPAD(Pin), PPout = PPAD(PoutT);
        const int ZPP   = 256 + PPin;
        const int O = K * cout;
        const bool idres = cfg[i].resid && !hasres[i];

        {   // zero Z halos once per layer
            int planes = NC * cout, per = ZPP - Pin, tot = planes * per;
            zero_halo_kernel<<<(tot + 255) / 256, 256, 0, stream>>>(Zchunk, planes, ZPP, Pin);
        }

        for (int ch = 0; ch < N / NC; ++ch) {
            const size_t n0 = (size_t)ch * NC;
            dim3 gg((Pin + 63) / 64, O / 64, NC);
            gcn_gemm_lds_kernel<<<gg, 256, 0, stream>>>(
                whg[i], Xcur + n0 * cin * PPin, Ychunk, b_gcn[i],
                O, cin, cpadG[i], Pin, PPin,
                (long long)cin * PPin, (long long)O * PPin);

            // aggregation on WMMA: (T/5)*(cout/16) units, 8 waves per block
            const int nTiles = cout / 16;
            const int units  = (Tcur / 5) * nTiles;
            dim3 ga(units / 8, NC);
            agg_wmma_kernel<<<ga, 256, 0, stream>>>(
                Ychunk, Aibuf + i * 1875, g1[i], be1[i], Zchunk,
                cout, PPin, ZPP, nTiles);

            const float* resBase = nullptr; long long res_ns = 0;
            if (hasres[i]) {
                dim3 gr((PoutT + 15) / 16, cout / 16, NC);
                res_gemm_wmma_kernel<<<gr, 32, 0, stream>>>(
                    whr[i], Xcur + n0 * cin * PPin, Rchunk,
                    b_res[i], g_res[i], be_res[i],
                    cout, cin, PoutT, PPout, PPin, V, s,
                    (long long)cin * PPin, (long long)cout * PPout);
                resBase = Rchunk; res_ns = (long long)cout * PPout;
            } else if (idres) {
                resBase = Xcur + n0 * cin * PPin; res_ns = (long long)cin * PPin;
            }

            dim3 gt((PoutT + 63) / 64, cout / 64, NC);
            tcn_gemm_lds_kernel<<<gt, 256, 0, stream>>>(
                wht[i], Zchunk, Xnext + n0 * cout * PPout, resBase,
                b_tcn[i], g2[i], be2[i],
                cout, cout, PoutT, PPout, ZPP, V, s,
                (long long)cout * PPout, res_ns);
        }
        float* tmp = Xcur; Xcur = Xnext; Xnext = tmp;
        Tcur = Tout;
    }

    pool_kernel<<<N, 256, 0, stream>>>(Xcur, pooled, 256, Tcur * V, PPAD(Tcur * V));
    fc_kernel<<<(N * 60 + 255) / 256, 256, 0, stream>>>(
        pooled, w_fc, b_fc, (float*)d_out, N, 256, 60);
}
